// NormalGuidedDeformable1DFeatureAggregator_82437602279994
// MI455X (gfx1250) — matrix operand order repaired and checked by reading, not compiled, not run on hardware
//
#include <hip/hip_runtime.h>
#include <hip/hip_bf16.h>

// ---------------------------------------------------------------------------
// NormalGuidedDeformable1DFeatureAggregator for MI455X (gfx1250, wave32)
//
// Bandwidth-bound (~4.3 GFLOP vs ~110 MB HBM traffic -> ~5us at 23.3 TB/s).
// All GEMMs run through v_wmma_f32_16x16x32_f16 (f16 in, f32 accumulate).
// v2 changes:
//  * weights zero-padded to N%16==0 at conversion -> no per-iteration
//    cndmask masking in the WMMA loop (loads + wmma only)
//  * one wave owns a 16-row M tile, preloads ALL A fragments for the full K
//    into registers, loops N tiles -> A read exactly once from HBM
//  * wc/wg/kpc/kpg fused into one (192,128) weight matrix sharing match_n
// ---------------------------------------------------------------------------

typedef _Float16 f16;
typedef __attribute__((ext_vector_type(16))) _Float16 v16h;
typedef __attribute__((ext_vector_type(8)))  _Float16 v8h;
typedef __attribute__((ext_vector_type(4)))  _Float16 v4h;
typedef __attribute__((ext_vector_type(8)))  float    v8f;

#define BSZ  2
#define CCH  128
#define HH   80
#define WW   160
#define HWD  (HH * WW)          // 12800
#define NPIX (BSZ * HWD)        // 25600
#define NPTS 9
#define NGRP 8
#define EPSV 1e-5f

// fused match-side GEMM output layout (all sections zero-padded to 16)
#define NFUSE 192
#define WOFF0 0                 // wc logits   [0,72)  pad to 80
#define WOFF1 80                // wg logits   [80,152) pad to 160
#define OOFF0 160               // kpc offsets [160,169) pad to 176
#define OOFF1 176               // kpg offsets [176,185) pad to 192

// ---------------------------------------------------------------------------
// small utility kernels
// ---------------------------------------------------------------------------
__global__ void cvt_f32_f16(const float* __restrict__ src, f16* __restrict__ dst, int n) {
    int i = blockIdx.x * 256 + threadIdx.x;
    if (i < n) dst[i] = (f16)src[i];
}
__global__ void fill_f16(f16* __restrict__ dst, int n) {
    int i = blockIdx.x * 256 + threadIdx.x;
    if (i < n) dst[i] = (f16)0.0f;
}
__global__ void fill_f32(float* __restrict__ dst, int n) {
    int i = blockIdx.x * 256 + threadIdx.x;
    if (i < n) dst[i] = 0.0f;
}
__global__ void copy_f32(const float* __restrict__ src, float* __restrict__ dst, int n) {
    int i = blockIdx.x * 256 + threadIdx.x;
    if (i < n) dst[i] = src[i];
}

// ---------------------------------------------------------------------------
// Channel LayerNorm of a (BS,C,H,W) tensor -> pixel-major (BS*HW, C) f16.
// 128-pixel tile per block; phase 1: coalesced channel-strided reads with
// per-thread (=per-pixel) mean/var; phase 2: per-thread (=per-channel)
// normalized writes, coalesced pixel-major.  LDS pitch 132 avoids bank
// conflicts on the transposed phase-2 reads.
// ---------------------------------------------------------------------------
__global__ __launch_bounds__(128) void layernorm_k(const float* __restrict__ in,
                                                   const float* __restrict__ gma,
                                                   const float* __restrict__ bta,
                                                   f16* __restrict__ out) {
    __shared__ f16  tile[CCH * 132];
    __shared__ float smean[128];
    __shared__ float srinv[128];

    const int  t       = threadIdx.x;
    const long pixbase = (long)blockIdx.x * 128;
    const long m       = pixbase + t;                 // global pixel id
    const int  b       = (int)(m / HWD);
    const int  pix     = (int)(m - (long)b * HWD);
    const float* src   = in + (long)b * CCH * HWD + pix;

    float s = 0.f, s2 = 0.f;
    for (int c = 0; c < CCH; ++c) {
        float v = src[(long)c * HWD];
        s += v; s2 += v * v;
        tile[c * 132 + t] = (f16)v;
    }
    float mean = s * (1.0f / CCH);
    float var  = s2 * (1.0f / CCH) - mean * mean;
    smean[t] = mean;
    srinv[t] = rsqrtf(var + EPSV);
    __syncthreads();

    const float gt = gma[t];                          // t == channel in phase 2
    const float bt = bta[t];
    for (int p = 0; p < 128; ++p) {
        float v = (float)tile[t * 132 + p];
        float o = (v - smean[p]) * srinv[p] * gt + bt;
        out[(pixbase + p) * CCH + t] = (f16)o;
    }
}

// ---------------------------------------------------------------------------
// WMMA GEMM:  D(M,N) = A(M,K) x W(N,K)^T + bias.  A,W f16 (N,K padded to 16),
// f32 accumulate.  One wave per 16-row M tile: all A fragments for the full
// K preloaded in registers (K/32 x v16h), then loop over N tiles reusing
// them -> A is read exactly once from memory; weights stay cache-resident.
// Fragment construction follows CDNA5 ISA 7.12.2 lane layouts:
//   A: lane l (half = l>>4) holds row (l&15), K chunks {kb..kb+7, 16+kb..}
//   B: lane l holds column (l&15), K chunks {kb..kb+7, 16+kb..} (same as A)
//   C/D: lane l, vgpr r -> D[r + 8*half][l&15]
// MODE: 0 = f32 row-major, 1 = f16 row-major, 2 = f32 (bs,C,H,W) into d_out
// ---------------------------------------------------------------------------
template <int K, int MODE>
__global__ __launch_bounds__(256) void gemm_wmma_k(const f16* __restrict__ A,
                                                   const f16* __restrict__ Wt,
                                                   const float* __restrict__ bias,
                                                   void* __restrict__ out,
                                                   int Mtot, int N) {
    constexpr int KS = K / 32;
    const int  wave = threadIdx.x >> 5;
    const int  lane = threadIdx.x & 31;
    const long tm   = (long)blockIdx.x * 8 + wave;    // M tile index
    if (tm >= (Mtot >> 4)) return;                    // uniform per wave

    const int lhalf = lane >> 4;                      // 0 or 1
    const int l15   = lane & 15;
    const int kbase = lhalf ? 8 : 0;
    const int koffB = lhalf ? 8 : 0;

    // preload all A fragments for this M tile (row l15, full K)
    const f16* Arow = A + (tm * 16 + l15) * (long)K;
    v16h a[KS];
#pragma unroll
    for (int ks = 0; ks < KS; ++ks) {
        v8h alo = *(const v8h*)(Arow + ks * 32 + kbase);
        v8h ahi = *(const v8h*)(Arow + ks * 32 + kbase + 16);
#pragma unroll
        for (int j = 0; j < 8; ++j) { a[ks][j] = alo[j]; a[ks][8 + j] = ahi[j]; }
    }

    const int tilesN = N >> 4;
    for (int tn = 0; tn < tilesN; ++tn) {
        const int  ncol = tn * 16 + l15;
        const f16* Wrow = Wt + (long)ncol * K;
        v8f acc = {};
#pragma unroll
        for (int ks = 0; ks < KS; ++ks) {
            v8h b0 = *(const v8h*)(Wrow + ks * 32 + koffB);
            v8h b1 = *(const v8h*)(Wrow + ks * 32 + koffB + 16);
            v16h bm;
#pragma unroll
            for (int j = 0; j < 8; ++j) { bm[j] = b0[j]; bm[8 + j] = b1[j]; }
            acc = __builtin_amdgcn_wmma_f32_16x16x32_f16(
                /*neg_a=*/false, a[ks], /*neg_b=*/false, bm,
                /*c_mod=*/(short)0, acc, /*reuse_a=*/false, /*reuse_b=*/false);
        }
        const float bv = bias[ncol];
#pragma unroll
        for (int r = 0; r < 8; ++r) {
            long  mg = tm * 16 + r + (lhalf ? 8 : 0);
            float v  = acc[r] + bv;
            if constexpr (MODE == 0) {
                ((float*)out)[mg * N + ncol] = v;
            } else if constexpr (MODE == 1) {
                ((f16*)out)[mg * N + ncol] = (f16)v;
            } else {
                int b   = (int)(mg / HWD);
                int pix = (int)(mg - (long)b * HWD);
                ((float*)out)[(long)b * CCH * HWD + (long)ncol * HWD + pix] = v;
            }
        }
    }
}

// ---------------------------------------------------------------------------
// Deformable 1D aggregation.  One wave per pixel; lane owns 4 contiguous
// channels (=> a single group g = lane>>2).  Lanes 0..17 compute + emit the
// keypoints, broadcast to the wave via __shfl.  Bilinear corners use clamped
// addresses with zeroed weights outside, so loads are always in-bounds and
// all wave-uniform branches stay convergent.
// grid.y = branch (0: context, 1: geometric)
// ---------------------------------------------------------------------------
__global__ __launch_bounds__(256) void aggregate_k(const f16* __restrict__ ctx_val,
                                                   const f16* __restrict__ geo_val,
                                                   const float* __restrict__ matchout,
                                                   const float* __restrict__ anc_c,
                                                   const float* __restrict__ anc_g,
                                                   const float* __restrict__ mask,
                                                   f16* __restrict__ cat,
                                                   float* __restrict__ kp_c,
                                                   float* __restrict__ kp_g) {
    const int  wave = threadIdx.x >> 5;
    const int  lane = threadIdx.x & 31;
    const long m    = (long)blockIdx.x * 8 + wave;
    if (m >= NPIX) return;
    const int branch = blockIdx.y;

    const f16*   feat = branch ? geo_val : ctx_val;
    const float* wmap = matchout + m * NFUSE + (branch ? WOFF1 : WOFF0);
    const float* offP = matchout + m * NFUSE + (branch ? OOFF1 : OOFF0);
    const float* anc  = branch ? anc_g : anc_c;
    float*       kp   = branch ? kp_g : kp_c;

    const int b   = (int)(m / HWD);
    const int pix = (int)(m - (long)b * HWD);

    // --- keypoints (lanes 0..17: lane = 2*p + comp) ---
    float kpv = 0.f;
    if (lane < 2 * NPTS) {
        int   p    = lane >> 1;
        int   comp = lane & 1;
        float a    = anc[m * (2 * NPTS) + lane];
        kpv        = comp ? a : (a + offP[p]);
        kp[m * (2 * NPTS) + lane] = kpv;
    }

    // --- group softmax over P for this lane's group ---
    const int g = lane >> 2;                          // c0 = 4*lane -> group c0/16
    float wt[NPTS];
    float mx = -3.4e38f;
#pragma unroll
    for (int p = 0; p < NPTS; ++p) {
        wt[p] = wmap[p * NGRP + g];
        mx    = fmaxf(mx, wt[p]);
    }
    float sum = 0.f;
#pragma unroll
    for (int p = 0; p < NPTS; ++p) { wt[p] = __expf(wt[p] - mx); sum += wt[p]; }
    const float rs = 1.f / sum;

    // --- bilinear gather + weighted sum over points ---
    float acc0 = 0.f, acc1 = 0.f, acc2 = 0.f, acc3 = 0.f;
    const long featbase = (long)b * HWD;
#pragma unroll
    for (int p = 0; p < NPTS; ++p) {
        float kx  = __shfl(kpv, 2 * p);
        float ky  = __shfl(kpv, 2 * p + 1);
        float x   = kx * WW - 0.5f;
        float y   = ky * HH - 0.5f;
        float x0f = floorf(x), y0f = floorf(y);
        float dx  = x - x0f,   dy  = y - y0f;
        int   x0  = (int)x0f,  y0  = (int)y0f;
        float wp  = wt[p] * rs;
#pragma unroll
        for (int cnr = 0; cnr < 4; ++cnr) {
            int   xi = x0 + (cnr & 1);
            int   yi = y0 + (cnr >> 1);
            float cw = ((cnr & 1) ? dx : 1.f - dx) * ((cnr >> 1) ? dy : 1.f - dy);
            bool  valid = (xi >= 0) && (xi < WW) && (yi >= 0) && (yi < HH);
            int   xc = min(max(xi, 0), WW - 1);
            int   yc = min(max(yi, 0), HH - 1);
            float wgt = valid ? cw * wp : 0.f;
            v4h hv = *(const v4h*)(feat + (featbase + (long)yc * WW + xc) * CCH + 4 * lane);
            acc0 += wgt * (float)hv[0];
            acc1 += wgt * (float)hv[1];
            acc2 += wgt * (float)hv[2];
            acc3 += wgt * (float)hv[3];
        }
    }

    const float mk  = branch ? 1.f : mask[(long)b * HWD + pix];
    f16* dst = cat + m * (2 * CCH) + branch * CCH + 4 * lane;
    dst[0] = (f16)(acc0 * mk);
    dst[1] = (f16)(acc1 * mk);
    dst[2] = (f16)(acc2 * mk);
    dst[3] = (f16)(acc3 * mk);
}

// ---------------------------------------------------------------------------
// Host-side orchestration
// ---------------------------------------------------------------------------
extern "C" void kernel_launch(void* const* d_in, const int* in_sizes, int n_in,
                              void* d_out, int out_size, void* d_ws, size_t ws_size,
                              hipStream_t stream) {
    (void)in_sizes; (void)n_in; (void)out_size; (void)ws_size;

    const float* mask    = (const float*)d_in[0];
    const float* match   = (const float*)d_in[1];
    const float* context = (const float*)d_in[2];
    const float* geom    = (const float*)d_in[3];
    const float* anc_c   = (const float*)d_in[4];
    const float* anc_g   = (const float*)d_in[5];
    const float* ln_g    = (const float*)d_in[6];
    const float* ln_b    = (const float*)d_in[7];
    const float* lnc_g   = (const float*)d_in[8];
    const float* lnc_b   = (const float*)d_in[9];
    const float* lng_g   = (const float*)d_in[10];
    const float* lng_b   = (const float*)d_in[11];
    const float* wc_w    = (const float*)d_in[12];
    const float* wc_b    = (const float*)d_in[13];
    const float* wg_w    = (const float*)d_in[14];
    const float* wg_b    = (const float*)d_in[15];
    const float* vc_w    = (const float*)d_in[16];
    const float* vc_b    = (const float*)d_in[17];
    const float* vg_w    = (const float*)d_in[18];
    const float* vg_b    = (const float*)d_in[19];
    const float* out_w   = (const float*)d_in[20];
    const float* out_b   = (const float*)d_in[21];
    const float* kpc_w   = (const float*)d_in[22];
    const float* kpc_b   = (const float*)d_in[23];
    const float* kpg_w   = (const float*)d_in[24];
    const float* kpg_b   = (const float*)d_in[25];

    // workspace carve-out (256B aligned)
    char*  ws = (char*)d_ws;
    size_t o  = 0;
    auto alloc = [&](size_t bytes) -> char* {
        char* p = ws + o;
        o += (bytes + 255) & ~(size_t)255;
        return p;
    };
    f16*   match_h  = (f16*)alloc((size_t)NPIX * CCH * 2);
    f16*   ctx_h    = (f16*)alloc((size_t)NPIX * CCH * 2);
    f16*   geo_h    = (f16*)alloc((size_t)NPIX * CCH * 2);
    f16*   vc_h     = (f16*)alloc((size_t)CCH * CCH * 2);
    f16*   vg_h     = (f16*)alloc((size_t)CCH * CCH * 2);
    f16*   packW_h  = (f16*)alloc((size_t)NFUSE * CCH * 2);   // fused wc|wg|kpc|kpg
    float* packB    = (float*)alloc((size_t)NFUSE * 4);       // fused bias
    f16*   outw_h   = (f16*)alloc((size_t)CCH * 2 * CCH * 2);
    f16*   ctxval_h = (f16*)alloc((size_t)NPIX * CCH * 2);
    f16*   geoval_h = (f16*)alloc((size_t)NPIX * CCH * 2);
    float* matchout = (float*)alloc((size_t)NPIX * NFUSE * 4);
    f16*   cat_h    = (f16*)alloc((size_t)NPIX * 2 * CCH * 2);

    auto cvt = [&](const float* s, f16* d, int n) {
        cvt_f32_f16<<<(n + 255) / 256, 256, 0, stream>>>(s, d, n);
    };

    // 1) weights -> f16 (fused match-side weights zero-padded to N%16==0)
    fill_f16<<<(NFUSE * CCH + 255) / 256, 256, 0, stream>>>(packW_h, NFUSE * CCH);
    fill_f32<<<1, 256, 0, stream>>>(packB, NFUSE);
    cvt(vc_w,  vc_h,  CCH * CCH);
    cvt(vg_w,  vg_h,  CCH * CCH);
    cvt(out_w, outw_h, CCH * 2 * CCH);
    cvt(wc_w,  packW_h + (size_t)WOFF0 * CCH, NGRP * NPTS * CCH);
    cvt(wg_w,  packW_h + (size_t)WOFF1 * CCH, NGRP * NPTS * CCH);
    cvt(kpc_w, packW_h + (size_t)OOFF0 * CCH, NPTS * CCH);
    cvt(kpg_w, packW_h + (size_t)OOFF1 * CCH, NPTS * CCH);
    copy_f32<<<1, 256, 0, stream>>>(wc_b,  packB + WOFF0, NGRP * NPTS);
    copy_f32<<<1, 256, 0, stream>>>(wg_b,  packB + WOFF1, NGRP * NPTS);
    copy_f32<<<1, 256, 0, stream>>>(kpc_b, packB + OOFF0, NPTS);
    copy_f32<<<1, 256, 0, stream>>>(kpg_b, packB + OOFF1, NPTS);

    // 2) layernorms -> pixel-major f16
    layernorm_k<<<NPIX / 128, 128, 0, stream>>>(match,   ln_g,  ln_b,  match_h);
    layernorm_k<<<NPIX / 128, 128, 0, stream>>>(context, lnc_g, lnc_b, ctx_h);
    layernorm_k<<<NPIX / 128, 128, 0, stream>>>(geom,    lng_g, lng_b, geo_h);

    // 3) WMMA GEMMs (1600 M tiles, 8 waves per 256-thread block -> 200 blocks)
    const int gblocks = (NPIX / 16 + 7) / 8;
    gemm_wmma_k<CCH, 1><<<gblocks, 256, 0, stream>>>(ctx_h, vc_h, vc_b,
                                                     ctxval_h, NPIX, CCH);
    gemm_wmma_k<CCH, 1><<<gblocks, 256, 0, stream>>>(geo_h, vg_h, vg_b,
                                                     geoval_h, NPIX, CCH);
    gemm_wmma_k<CCH, 0><<<gblocks, 256, 0, stream>>>(match_h, packW_h, packB,
                                                     matchout, NPIX, NFUSE);

    // 4) deformable aggregation (+ keypoint outputs straight into d_out)
    float* outp = (float*)d_out;
    float* kp_c = outp + (size_t)BSZ * CCH * HWD;
    float* kp_g = kp_c + (size_t)BSZ * HWD * NPTS * 2;
    aggregate_k<<<dim3(NPIX / 8, 2), 256, 0, stream>>>(
        ctxval_h, geoval_h, matchout, anc_c, anc_g, mask, cat_h, kp_c, kp_g);

    // 5) output projection: (NPIX,256) x (128,256)^T -> (bs,C,H,W) f32 in d_out
    gemm_wmma_k<2 * CCH, 2><<<gblocks, 256, 0, stream>>>(cat_h, outw_h, out_b,
                                                         outp, NPIX, CCH);
}
